// PointNet2Classifier_4011499454876
// MI455X (gfx1250) — matrix operand order repaired and checked by reading, not compiled, not run on hardware
//
#include <hip/hip_runtime.h>
#include <hip/hip_bf16.h>

typedef __attribute__((ext_vector_type(16))) _Float16 v16h;
typedef __attribute__((ext_vector_type(8)))  _Float16 v8h;
typedef __attribute__((ext_vector_type(8)))  float    v8f;

// ---------------------------------------------------------------------------
// Farthest point sampling: one workgroup (256 threads) per batch.
// dist[] kept resident in LDS; npoint serial steps of (update-dist, argmax).
// Emits new_xyz directly (gather fused).
// ---------------------------------------------------------------------------
__global__ void fps_kernel(const float* __restrict__ xyz, int N, int npoint,
                           float* __restrict__ new_xyz)
{
    const int b = blockIdx.x;
    const float* X = xyz + (long)b * N * 3;
    __shared__ float dist[4096];
    __shared__ float rv[256];
    __shared__ int   ri[256];
    __shared__ float cx, cy, cz;
    __shared__ int   far_s;

    for (int i = threadIdx.x; i < N; i += blockDim.x) dist[i] = 1e10f;
    if (threadIdx.x == 0) far_s = 0;
    __syncthreads();

    for (int it = 0; it < npoint; ++it) {
        if (threadIdx.x == 0) {
            int f = far_s;
            float x = X[f*3+0], y = X[f*3+1], z = X[f*3+2];
            cx = x; cy = y; cz = z;
            float* o = new_xyz + ((long)b * npoint + it) * 3;
            o[0] = x; o[1] = y; o[2] = z;
        }
        __syncthreads();
        float bestv = -1.0f; int besti = 0;
        for (int i = threadIdx.x; i < N; i += blockDim.x) {
            float dx = X[i*3+0] - cx;
            float dy = X[i*3+1] - cy;
            float dz = X[i*3+2] - cz;
            float d  = dx*dx + dy*dy + dz*dz;
            float m  = fminf(dist[i], d);
            dist[i]  = m;
            if (m > bestv) { bestv = m; besti = i; }
        }
        rv[threadIdx.x] = bestv; ri[threadIdx.x] = besti;
        __syncthreads();
        for (int s = 128; s > 0; s >>= 1) {
            if ((int)threadIdx.x < s) {
                float ov = rv[threadIdx.x + s]; int oi = ri[threadIdx.x + s];
                if (ov > rv[threadIdx.x] ||
                    (ov == rv[threadIdx.x] && oi < ri[threadIdx.x])) {
                    rv[threadIdx.x] = ov; ri[threadIdx.x] = oi;
                }
            }
            __syncthreads();
        }
        if (threadIdx.x == 0) far_s = ri[0];
        __syncthreads();
    }
}

// ---------------------------------------------------------------------------
// Ball query: one thread per query center. Collect up to nsample neighbor
// indices in ascending order (== reference sort-then-truncate), pad with the
// first hit.
// ---------------------------------------------------------------------------
__global__ void ball_query_kernel(const float* __restrict__ xyz,
                                  const float* __restrict__ new_xyz,
                                  int* __restrict__ out_idx,
                                  int total, int N, int S, int nsample, float r2)
{
    int t = blockIdx.x * blockDim.x + threadIdx.x;
    if (t >= total) return;
    int b = t / S;
    const float* X = xyz + (long)b * N * 3;
    float qx = new_xyz[t*3+0], qy = new_xyz[t*3+1], qz = new_xyz[t*3+2];
    int* out = out_idx + (long)t * nsample;
    int cnt = 0, first = 0;
    for (int i = 0; i < N && cnt < nsample; ++i) {
        float dx = X[i*3+0] - qx;
        float dy = X[i*3+1] - qy;
        float dz = X[i*3+2] - qz;
        float d2 = dx*dx + dy*dy + dz*dz;
        if (d2 <= r2) {
            if (cnt == 0) first = i;
            out[cnt++] = i;
        }
    }
    for (; cnt < nsample; ++cnt) out[cnt] = first;
}

// ---------------------------------------------------------------------------
// Grouping kernels: build the zero-padded f16 activation matrix (rows x Kpad).
// ---------------------------------------------------------------------------
__global__ void group_xyz_kernel(const float* __restrict__ xyz,
                                 const float* __restrict__ nxyz,
                                 const int* __restrict__ idx,
                                 _Float16* __restrict__ A,
                                 int S, int ns, int N, int Kpad, long rows)
{
    long t = (long)blockIdx.x * blockDim.x + threadIdx.x;
    if (t >= rows) return;
    long g = t / ns;
    int  b = (int)(g / S);
    int  pi = idx[t];
    const float* p = xyz  + ((long)b * N + pi) * 3;
    const float* q = nxyz + g * 3;
    _Float16* row = A + t * Kpad;
    row[0] = (_Float16)(p[0] - q[0]);
    row[1] = (_Float16)(p[1] - q[1]);
    row[2] = (_Float16)(p[2] - q[2]);
    for (int k = 3; k < Kpad; ++k) row[k] = (_Float16)0.0f;
}

__global__ void group_feat_kernel(const float* __restrict__ xyz,
                                  const float* __restrict__ nxyz,
                                  const int* __restrict__ idx,
                                  const float* __restrict__ feat, int C,
                                  _Float16* __restrict__ A,
                                  int S, int ns, int N, int Kpad, long rows)
{
    long t = (long)blockIdx.x * blockDim.x + threadIdx.x;
    if (t >= rows) return;
    long g = t / ns;
    int  b = (int)(g / S);
    int  pi = idx[t];
    const float* p = xyz  + ((long)b * N + pi) * 3;
    const float* q = nxyz + g * 3;
    const float* f = feat + ((long)b * N + pi) * C;
    _Float16* row = A + t * Kpad;
    row[0] = (_Float16)(p[0] - q[0]);
    row[1] = (_Float16)(p[1] - q[1]);
    row[2] = (_Float16)(p[2] - q[2]);
    for (int c = 0; c < C; ++c) row[3 + c] = (_Float16)f[c];
    for (int k = 3 + C; k < Kpad; ++k) row[k] = (_Float16)0.0f;
}

__global__ void group_all_kernel(const float* __restrict__ xyz,
                                 const float* __restrict__ feat, int C,
                                 _Float16* __restrict__ A, int Kpad, long rows)
{
    long t = (long)blockIdx.x * blockDim.x + threadIdx.x;
    if (t >= rows) return;
    const float* p = xyz  + t * 3;
    const float* f = feat + t * C;
    _Float16* row = A + t * Kpad;
    row[0] = (_Float16)p[0];
    row[1] = (_Float16)p[1];
    row[2] = (_Float16)p[2];
    for (int c = 0; c < C; ++c) row[3 + c] = (_Float16)f[c];
    for (int k = 3 + C; k < Kpad; ++k) row[k] = (_Float16)0.0f;
}

// ---------------------------------------------------------------------------
// Max pool over the group (nsample) dimension. Optional f32 and f16 outputs.
// ---------------------------------------------------------------------------
__global__ void maxpool_kernel(const _Float16* __restrict__ X, int ld,
                               int ns, int C, long rows,
                               float* __restrict__ outf,
                               _Float16* __restrict__ outh, int ldh)
{
    long t = (long)blockIdx.x * blockDim.x + threadIdx.x;
    if (t >= rows * (long)C) return;
    long g = t / C;
    int  c = (int)(t % C);
    float m = -3.0e38f;
    const _Float16* base = X + (g * ns) * (long)ld + c;
    for (int j = 0; j < ns; ++j)
        m = fmaxf(m, (float)base[(long)j * ld]);
    if (outf) outf[g * (long)C + c] = m;
    if (outh) outh[g * (long)ldh + c] = (_Float16)m;
}

// ---------------------------------------------------------------------------
// Weight prep: W (N x K, f32) -> Wt (Npad x Kpad, f16, zero padded), and
// per-channel BN+bias fold into scale/shift (pads get scale=shift=0).
// ---------------------------------------------------------------------------
__global__ void prep_weights_kernel(const float* __restrict__ W,
                                    const float* __restrict__ bias,
                                    const float* __restrict__ gamma,
                                    const float* __restrict__ beta,
                                    const float* __restrict__ mu,
                                    const float* __restrict__ var,
                                    _Float16* __restrict__ Wt,
                                    float* __restrict__ scale,
                                    float* __restrict__ shift,
                                    int N, int K, int Npad, int Kpad, int has_bn)
{
    int i = blockIdx.x * blockDim.x + threadIdx.x;
    if (i < Npad * Kpad) {
        int n = i / Kpad, k = i % Kpad;
        float w = (n < N && k < K) ? W[(long)n * K + k] : 0.0f;
        Wt[(long)n * Kpad + k] = (_Float16)w;
    }
    if (i < Npad) {
        float sc = 0.0f, sh = 0.0f;
        if (i < N) {
            if (has_bn) {
                float s = gamma[i] * rsqrtf(var[i] + 1e-5f);
                sc = s;
                sh = s * (bias[i] - mu[i]) + beta[i];
            } else {
                sc = 1.0f;
                sh = bias[i];
            }
        }
        scale[i] = sc;
        shift[i] = sh;
    }
}

// ---------------------------------------------------------------------------
// WMMA GEMM: D(M x N) = A(M x K) * Wt(N x K)^T, fused scale/shift/ReLU.
// A row-major f16 (lda = Kpad, multiple of 32, zero padded).
// Wt row-major f16: row n = weights of output channel n (contiguous K).
// One wave computes a 16x64 output tile: the A fragment is loaded once per
// K-step and reused across 4 independent back-to-back v_wmma_f32_16x16x32_f16
// (XDL-friendly, 4x less A traffic). Tile index is wave-uniform so EXEC is
// all-ones at every v_wmma. N must be a multiple of 64.
// ---------------------------------------------------------------------------
__global__ void wmma_gemm_kernel(const _Float16* __restrict__ A, int lda,
                                 const _Float16* __restrict__ Wt, int ldw,
                                 const float* __restrict__ scale,
                                 const float* __restrict__ shift,
                                 _Float16* __restrict__ Oh, int ldo,
                                 float* __restrict__ Of, int ldof, int Nreal,
                                 int M, int N, int K, int relu)
{
    const int lane = threadIdx.x & 31;
    const int gw   = blockIdx.x * (blockDim.x >> 5) + (threadIdx.x >> 5);
    const int tilesN4 = N >> 6;               // 64-wide column groups
    const int tilesM  = M >> 4;
    if (gw >= tilesM * tilesN4) return;
    const int tm = gw / tilesN4;
    const int tn = gw - tm * tilesN4;
    const int row0 = tm << 4;
    const int col0 = tn << 6;
    const bool hi  = lane >= 16;
    const int  l15 = lane & 15;

    const long arow    = (long)(row0 + l15) * lda + (hi ? 16 : 0);
    const long brow    = (long)(col0 + l15) * ldw + (hi ? 16 : 0);
    const long bstride = (long)16 * ldw;

    v8f acc[4] = {{}, {}, {}, {}};
    for (int k0 = 0; k0 < K; k0 += 32) {
        const _Float16* ap = A + arow + k0;
        v8h a0 = *(const v8h*)(ap);
        v8h a1 = *(const v8h*)(ap + 8);
        v16h afrag = __builtin_shufflevector(a0, a1,
            0,1,2,3,4,5,6,7,8,9,10,11,12,13,14,15);
        const _Float16* bp = Wt + brow + k0;
#pragma unroll
        for (int j = 0; j < 4; ++j) {
            v16h bfrag = *(const v16h*)(bp + (long)j * bstride);
            acc[j] = __builtin_amdgcn_wmma_f32_16x16x32_f16(
                /*neg_a=*/false, afrag, /*neg_b=*/false, bfrag,
                /*c_mod=*/(short)0, acc[j], /*reuse_a=*/false, /*reuse_b=*/false);
        }
    }

#pragma unroll
    for (int j = 0; j < 4; ++j) {
        const int n = col0 + 16 * j + l15;
        const float scv = scale[n];
        const float shv = shift[n];
#pragma unroll
        for (int v = 0; v < 8; ++v) {
            int m = row0 + v + (hi ? 8 : 0);
            float f = acc[j][v] * scv + shv;
            if (relu) f = fmaxf(f, 0.0f);
            if (Oh) Oh[(long)m * ldo + n] = (_Float16)f;
            if (Of && n < Nreal) Of[(long)m * ldof + n] = f;
        }
    }
}

// ---------------------------------------------------------------------------
// Host-side orchestration
// ---------------------------------------------------------------------------
static inline size_t align_up(size_t x, size_t a) { return (x + a - 1) & ~(a - 1); }

struct LayerDesc { int N, K, Npad, Kpad, wi, has_bn; };

static void launch_gemm(const _Float16* A, int lda, const _Float16* Wt, int ldw,
                        const float* sc, const float* sh,
                        _Float16* Oh, int ldo, float* Of, int ldof, int Nreal,
                        int M, int N, int K, int relu, hipStream_t stream)
{
    long tiles = (long)(M / 16) * (N / 64);  // one wave per 16x64 tile
    long blocks = (tiles + 7) / 8;           // 8 waves / 256-thread block
    wmma_gemm_kernel<<<dim3((unsigned)blocks), 256, 0, stream>>>(
        A, lda, Wt, ldw, sc, sh, Oh, ldo, Of, ldof, Nreal, M, N, K, relu);
}

extern "C" void kernel_launch(void* const* d_in, const int* in_sizes, int n_in,
                              void* d_out, int out_size, void* d_ws, size_t ws_size,
                              hipStream_t stream)
{
    (void)in_sizes; (void)n_in; (void)out_size; (void)ws_size;
    const int B = 32, N0 = 4096;
    const int S1 = 512, NS1 = 32;
    const int S2 = 128, NS2 = 64;

    const float* points = (const float*)d_in[0];

    // input index -> layer table  (W, b, g, beta, mu, var per BN layer)
    static const LayerDesc L[12] = {
        {  64,    3,   64,   32,  1, 1},  // SA1.0
        {  64,   64,   64,   64,  7, 1},  // SA1.1
        { 128,   64,  128,   64, 13, 1},  // SA1.2
        { 128,  131,  128,  160, 19, 1},  // SA2.0
        { 128,  128,  128,  128, 25, 1},  // SA2.1
        { 256,  128,  256,  128, 31, 1},  // SA2.2
        { 256,  259,  256,  288, 37, 1},  // SA3.0
        { 512,  256,  512,  256, 43, 1},  // SA3.1
        {1024,  512, 1024,  512, 49, 1},  // SA3.2
        { 512, 1024,  512, 1024, 55, 1},  // FC1
        { 256,  512,  256,  512, 61, 1},  // FC2
        {  40,  256,   64,  256, 67, 0},  // FC3 (N padded to 64)
    };

    // ---- carve workspace ----
    char* ws = (char*)d_ws;
    size_t off = 0;
    auto carve = [&](size_t bytes) -> void* {
        void* p = ws + off;
        off = align_up(off + bytes, 256);
        return p;
    };

    float* nxyz1   = (float*)carve((size_t)B * S1 * 3 * 4);
    float* nxyz2   = (float*)carve((size_t)B * S2 * 3 * 4);
    int*   bidx1   = (int*)  carve((size_t)B * S1 * NS1 * 4);
    int*   bidx2   = (int*)  carve((size_t)B * S2 * NS2 * 4);
    float* l1pts   = (float*)carve((size_t)B * S1 * 128 * 4);
    float* l2pts   = (float*)carve((size_t)B * S2 * 256 * 4);
    _Float16* bufA = (_Float16*)carve((size_t)134217728);
    _Float16* bufB = (_Float16*)carve((size_t)134217728);

    _Float16* Wt[12]; float* sc[12]; float* sh[12];
    for (int i = 0; i < 12; ++i) {
        Wt[i] = (_Float16*)carve((size_t)L[i].Npad * L[i].Kpad * 2);
        sc[i] = (float*)carve((size_t)L[i].Npad * 4);
        sh[i] = (float*)carve((size_t)L[i].Npad * 4);
    }

    // ---- weight prep (12 small kernels) ----
    for (int i = 0; i < 12; ++i) {
        const float* W  = (const float*)d_in[L[i].wi];
        const float* bb = (const float*)d_in[L[i].wi + 1];
        const float *g = nullptr, *be = nullptr, *mu = nullptr, *vr = nullptr;
        if (L[i].has_bn) {
            g  = (const float*)d_in[L[i].wi + 2];
            be = (const float*)d_in[L[i].wi + 3];
            mu = (const float*)d_in[L[i].wi + 4];
            vr = (const float*)d_in[L[i].wi + 5];
        }
        int tot = L[i].Npad * L[i].Kpad;
        prep_weights_kernel<<<(tot + 255) / 256, 256, 0, stream>>>(
            W, bb, g, be, mu, vr, Wt[i], sc[i], sh[i],
            L[i].N, L[i].K, L[i].Npad, L[i].Kpad, L[i].has_bn);
    }

    // ---- SA1 ----
    fps_kernel<<<B, 256, 0, stream>>>(points, N0, S1, nxyz1);
    {
        int total = B * S1;
        ball_query_kernel<<<(total + 255) / 256, 256, 0, stream>>>(
            points, nxyz1, bidx1, total, N0, S1, NS1, 0.2f * 0.2f);
    }
    long rows1 = (long)B * S1 * NS1;                       // 524288
    group_xyz_kernel<<<(unsigned)((rows1 + 255) / 256), 256, 0, stream>>>(
        points, nxyz1, bidx1, bufA, S1, NS1, N0, 32, rows1);
    launch_gemm(bufA, 32,  Wt[0], 32,  sc[0], sh[0], bufB, 64,  nullptr, 0, 0,
                (int)rows1, 64, 32, 1, stream);
    launch_gemm(bufB, 64,  Wt[1], 64,  sc[1], sh[1], bufA, 64,  nullptr, 0, 0,
                (int)rows1, 64, 64, 1, stream);
    launch_gemm(bufA, 64,  Wt[2], 64,  sc[2], sh[2], bufB, 128, nullptr, 0, 0,
                (int)rows1, 128, 64, 1, stream);
    {
        long rc = (long)B * S1;
        long tot = rc * 128;
        maxpool_kernel<<<(unsigned)((tot + 255) / 256), 256, 0, stream>>>(
            bufB, 128, NS1, 128, rc, l1pts, nullptr, 0);
    }

    // ---- SA2 ----
    fps_kernel<<<B, 256, 0, stream>>>(nxyz1, S1, S2, nxyz2);
    {
        int total = B * S2;
        ball_query_kernel<<<(total + 255) / 256, 256, 0, stream>>>(
            nxyz1, nxyz2, bidx2, total, S1, S2, NS2, 0.4f * 0.4f);
    }
    long rows2 = (long)B * S2 * NS2;                       // 262144
    group_feat_kernel<<<(unsigned)((rows2 + 255) / 256), 256, 0, stream>>>(
        nxyz1, nxyz2, bidx2, l1pts, 128, bufA, S2, NS2, S1, 160, rows2);
    launch_gemm(bufA, 160, Wt[3], 160, sc[3], sh[3], bufB, 128, nullptr, 0, 0,
                (int)rows2, 128, 160, 1, stream);
    launch_gemm(bufB, 128, Wt[4], 128, sc[4], sh[4], bufA, 128, nullptr, 0, 0,
                (int)rows2, 128, 128, 1, stream);
    launch_gemm(bufA, 128, Wt[5], 128, sc[5], sh[5], bufB, 256, nullptr, 0, 0,
                (int)rows2, 256, 128, 1, stream);
    {
        long rc = (long)B * S2;
        long tot = rc * 256;
        maxpool_kernel<<<(unsigned)((tot + 255) / 256), 256, 0, stream>>>(
            bufB, 256, NS2, 256, rc, l2pts, nullptr, 0);
    }

    // ---- SA3 (group all) ----
    long rows3 = (long)B * S2;                             // 4096
    group_all_kernel<<<(unsigned)((rows3 + 255) / 256), 256, 0, stream>>>(
        nxyz2, l2pts, 256, bufA, 288, rows3);
    launch_gemm(bufA, 288, Wt[6], 288, sc[6], sh[6], bufB, 256,  nullptr, 0, 0,
                (int)rows3, 256, 288, 1, stream);
    launch_gemm(bufB, 256, Wt[7], 256, sc[7], sh[7], bufA, 512,  nullptr, 0, 0,
                (int)rows3, 512, 256, 1, stream);
    launch_gemm(bufA, 512, Wt[8], 512, sc[8], sh[8], bufB, 1024, nullptr, 0, 0,
                (int)rows3, 1024, 512, 1, stream);
    {
        long tot = (long)B * 1024;
        maxpool_kernel<<<(unsigned)((tot + 255) / 256), 256, 0, stream>>>(
            bufB, 1024, S2, 1024, B, nullptr, bufA, 1024);
    }

    // ---- FC head ----
    launch_gemm(bufA, 1024, Wt[9],  1024, sc[9],  sh[9],  bufB, 512, nullptr, 0, 0,
                B, 512, 1024, 1, stream);
    launch_gemm(bufB, 512,  Wt[10], 512,  sc[10], sh[10], bufA, 256, nullptr, 0, 0,
                B, 256, 512, 1, stream);
    launch_gemm(bufA, 256,  Wt[11], 256,  sc[11], sh[11], nullptr, 0,
                (float*)d_out, 40, 40, B, 64, 256, 0, stream);
}
